// AttentionDecoder_85341000171980
// MI455X (gfx1250) — matrix operand order repaired and checked by reading, not compiled, not run on hardware
//
#include <hip/hip_runtime.h>
#include <hip/hip_bf16.h>
#include <math.h>

// ---- problem dims (match reference) ----
#define B_     32
#define T_     2048
#define D_     256
#define H_     256
#define A_     128
#define C_     4367
#define STEPS_ 22

typedef __attribute__((ext_vector_type(16))) __bf16 v16bf;
typedef __attribute__((ext_vector_type(8)))  float  v8f;

// ---------------- WMMA fragment loaders (contiguous k order, same in A and B, wave32) ----------------
// A (16x32, 16-bit): lane m=lane&15 holds row M=m; half g=lane>>4:
//   elem i -> K = k0 + 16g + i (16 contiguous)
__device__ __forceinline__ v16bf load_a_f32(const float* __restrict__ A, int lda,
                                            int row0, int k0, int lane) {
  int m = lane & 15, g = lane >> 4;
  const float* p = A + (size_t)(row0 + m) * (size_t)lda + (size_t)(k0 + 16 * g);
  v16bf a;
#pragma unroll
  for (int i = 0; i < 8; ++i) a[i] = (__bf16)p[i];
#pragma unroll
  for (int i = 0; i < 8; ++i) a[8 + i] = (__bf16)p[8 + i];
  return a;
}

// B (32x16, 16-bit), weights stored transposed as [N,K] row-major (K contiguous):
//   lane m holds column N=n0+m; half g: elem i -> K = k0 + 16g + i (16 contiguous bf16)
__device__ __forceinline__ v16bf load_b_bf16(const __bf16* __restrict__ Bt, int ldb,
                                             int n0, int k0, int lane, int Nmax) {
  int m = lane & 15, g = lane >> 4;
  int n = n0 + m; if (n > Nmax - 1) n = Nmax - 1;   // clamp ragged N (store is guarded)
  const __bf16* p = Bt + (size_t)n * (size_t)ldb + (size_t)(k0 + 16 * g);
  v16bf b;
#pragma unroll
  for (int i = 0; i < 16; ++i) b[i] = p[i];
  return b;
}

// ------- generic WMMA GEMM: C[M,N] = A[M,K](f32) * Bt[N,K](bf16)^T (+bias) -----------
// One wave computes a 16x64 strip: 4 accumulators reuse each A fragment (4 WMMA / A-load).
__global__ void gemm_wmma_bf16(const float* __restrict__ Amat, int lda,
                               const __bf16* __restrict__ Bt, int ldb,
                               float* __restrict__ Cmat, int ldc,
                               const float* __restrict__ bias,
                               int M, int N, int K) {
  int lane = threadIdx.x & 31;
  int wave = blockIdx.x * (blockDim.x >> 5) + (threadIdx.x >> 5);
  int mTiles  = (M + 15) >> 4;
  int nGroups = (N + 63) >> 6;
  if (wave >= mTiles * nGroups) return;         // whole-wave uniform exit: EXEC all-1 for WMMA
  int mt = wave / nGroups, ng = wave % nGroups;
  int row0 = mt << 4, col0 = ng << 6;

  v8f acc[4] = {};
  for (int k0 = 0; k0 < K; k0 += 32) {
    v16bf a = load_a_f32(Amat, lda, row0, k0, lane);
#pragma unroll
    for (int j = 0; j < 4; ++j) {
      v16bf b = load_b_bf16(Bt, ldb, col0 + 16 * j, k0, lane, N);
      acc[j] = __builtin_amdgcn_wmma_f32_16x16x32_bf16(false, a, false, b,
                                                       (short)0, acc[j], false, false);
    }
  }
  // C/D layout: lane n=lane&15 -> col, vgpr r -> row r + 8*(lane>>4)
  int n = lane & 15, g = lane >> 4;
#pragma unroll
  for (int j = 0; j < 4; ++j) {
    int col = col0 + 16 * j + n;
    if (col < N) {
      float bv = bias ? bias[col] : 0.0f;
#pragma unroll
      for (int r = 0; r < 8; ++r) {
        int row = row0 + r + 8 * g;
        if (row < M) Cmat[(size_t)row * (size_t)ldc + col] = acc[j][r] + bv;
      }
    }
  }
}

// ---------------- weight convert (f32 -> bf16), optional transpose to [N,K] -----------
__global__ void conv_bf16_k(const float* __restrict__ src, __bf16* __restrict__ dst,
                            int rows, int cols, int do_transpose) {
  int idx = blockIdx.x * blockDim.x + threadIdx.x;
  if (idx >= rows * cols) return;
  int r = idx / cols, c = idx % cols;
  __bf16 v = (__bf16)src[idx];
  if (do_transpose) dst[(size_t)c * rows + r] = v;
  else              dst[idx] = v;
}

__global__ void zero_k(float* __restrict__ p, int n) {
  int i = blockIdx.x * blockDim.x + threadIdx.x;
  if (i < n) p[i] = 0.0f;
}

// ------- fused attention: scores -> softmax -> context (one block per batch) ----------
// e[t] = v . tanh(xW[b,t,:] + hWh[b,:]);  att = softmax_T(e);  ctx[b,d] = sum_t x[b,t,d]*att[t]
__global__ void attn_ctx_k(const float* __restrict__ xW,   // [B,T,A]
                           const float* __restrict__ hWh,  // [B,A]
                           const float* __restrict__ v,    // [A]
                           const float* __restrict__ x,    // [B,T,D]
                           float* __restrict__ ctx) {      // [B,D]
  __shared__ float sv[A_];
  __shared__ float sh[A_];
  __shared__ float se[T_];
  __shared__ float red[256];
  int b = blockIdx.x, tid = threadIdx.x;      // blockDim = 256 = D_

#if defined(__gfx1250__)
  // CDNA5 async global->LDS copy for the small per-step vectors (no VGPR round-trip).
  if (tid < A_) {
    unsigned lv = (unsigned)(size_t)&sv[tid];
    unsigned lh = (unsigned)(size_t)&sh[tid];
    const float* gv = v + tid;
    const float* gh = hWh + (size_t)b * A_ + tid;
    asm volatile("global_load_async_to_lds_b32 %0, %1, off\n\t"
                 "global_load_async_to_lds_b32 %2, %3, off"
                 :: "v"(lv), "v"(gv), "v"(lh), "v"(gh) : "memory");
  }
  asm volatile("s_wait_asynccnt 0x0" ::: "memory");
#else
  if (tid < A_) { sv[tid] = v[tid]; sh[tid] = hWh[(size_t)b * A_ + tid]; }
#endif
  __syncthreads();

  // scores
  for (int t = tid; t < T_; t += 256) {
    const float* p = xW + ((size_t)b * T_ + t) * A_;
    float acc = 0.f;
#pragma unroll 4
    for (int a = 0; a < A_; ++a) acc += sv[a] * tanhf(p[a] + sh[a]);
    se[t] = acc;
  }
  __syncthreads();

  // softmax over T
  float m = -INFINITY;
  for (int t = tid; t < T_; t += 256) m = fmaxf(m, se[t]);
  red[tid] = m; __syncthreads();
  for (int s = 128; s > 0; s >>= 1) { if (tid < s) red[tid] = fmaxf(red[tid], red[tid + s]); __syncthreads(); }
  m = red[0]; __syncthreads();

  float sum = 0.f;
  for (int t = tid; t < T_; t += 256) { float e = __expf(se[t] - m); se[t] = e; sum += e; }
  red[tid] = sum; __syncthreads();
  for (int s = 128; s > 0; s >>= 1) { if (tid < s) red[tid] += red[tid + s]; __syncthreads(); }
  float inv = 1.0f / red[0];
  for (int t = tid; t < T_; t += 256) se[t] *= inv;   // normalized att stays in LDS
  __syncthreads();

  // context: thread d accumulates over T (x rows are lane-coalesced across d)
  const float* xp = x + (size_t)b * T_ * D_ + tid;
  float acc = 0.f;
  for (int t = 0; t < T_; t += 4) {
    __builtin_prefetch(xp + (size_t)(t + 32) * D_, 0, 1);  // global_prefetch_b8
    acc += xp[(size_t)(t + 0) * D_] * se[t + 0];
    acc += xp[(size_t)(t + 1) * D_] * se[t + 1];
    acc += xp[(size_t)(t + 2) * D_] * se[t + 2];
    acc += xp[(size_t)(t + 3) * D_] * se[t + 3];
  }
  ctx[b * D_ + tid] = acc;
}

// ---------------- GRU gate fusion: h <- (1-z)*n + z*h --------------------------------
__global__ void gru_gate_k(const float* __restrict__ gi, const float* __restrict__ gh,
                           float* __restrict__ h) {
  int idx = blockIdx.x * blockDim.x + threadIdx.x;
  if (idx >= B_ * H_) return;
  int b = idx / H_, j = idx % H_;
  const float* gib = gi + (size_t)b * 3 * H_;
  const float* ghb = gh + (size_t)b * 3 * H_;
  float r = 1.0f / (1.0f + __expf(-(gib[j] + ghb[j])));
  float z = 1.0f / (1.0f + __expf(-(gib[H_ + j] + ghb[H_ + j])));
  float n = tanhf(gib[2 * H_ + j] + r * ghb[2 * H_ + j]);
  h[idx] = (1.0f - z) * n + z * h[idx];
}

// -------------------------------------------------------------------------------------
extern "C" void kernel_launch(void* const* d_in, const int* in_sizes, int n_in,
                              void* d_out, int out_size, void* d_ws, size_t ws_size,
                              hipStream_t stream) {
  const float* x     = (const float*)d_in[0];
  const float* Wx    = (const float*)d_in[1];
  const float* Wh    = (const float*)d_in[2];
  const float* v     = (const float*)d_in[3];
  const float* W_ih  = (const float*)d_in[4];
  const float* W_hh  = (const float*)d_in[5];
  const float* b_ih  = (const float*)d_in[6];
  const float* b_hh  = (const float*)d_in[7];
  const float* W_cls = (const float*)d_in[8];
  const float* b_cls = (const float*)d_in[9];
  float* out = (float*)d_out;

  // workspace carve-up (~37 MB)
  float* xW  = (float*)d_ws;                 // [B*T, A]
  float* hWh = xW  + (size_t)B_ * T_ * A_;   // [B, A]
  float* ctx = hWh + (size_t)B_ * A_;        // [B, D]
  float* gi  = ctx + (size_t)B_ * D_;        // [B, 3H]
  float* gh  = gi  + (size_t)B_ * 3 * H_;    // [B, 3H]
  float* h   = gh  + (size_t)B_ * 3 * H_;    // [B, H]
  __bf16* WxT  = (__bf16*)(h + (size_t)B_ * H_);   // [A, D]  (transposed)
  __bf16* WhT  = WxT  + (size_t)A_ * D_;           // [A, H]  (transposed)
  __bf16* Wihb = WhT  + (size_t)A_ * H_;           // [3H, D] (already [N,K])
  __bf16* Whhb = Wihb + (size_t)3 * H_ * D_;       // [3H, H]
  __bf16* Wclb = Whhb + (size_t)3 * H_ * H_;       // [C, H]

  auto cdiv = [](int a, int b) { return (a + b - 1) / b; };
  auto gemm_blocks = [&](int M, int N) {            // 4 waves per 128-thread block
    int tiles = cdiv(M, 16) * cdiv(N, 64);
    return cdiv(tiles, 4);
  };

  // 1) weight conversions (bf16, [N,K] row-major)
  conv_bf16_k<<<cdiv(D_ * A_, 256), 256, 0, stream>>>(Wx,    WxT,  D_, A_, 1);
  conv_bf16_k<<<cdiv(H_ * A_, 256), 256, 0, stream>>>(Wh,    WhT,  H_, A_, 1);
  conv_bf16_k<<<cdiv(3 * H_ * D_, 256), 256, 0, stream>>>(W_ih, Wihb, 3 * H_, D_, 0);
  conv_bf16_k<<<cdiv(3 * H_ * H_, 256), 256, 0, stream>>>(W_hh, Whhb, 3 * H_, H_, 0);
  conv_bf16_k<<<cdiv(C_ * H_, 256), 256, 0, stream>>>(W_cls, Wclb, C_, H_, 0);
  zero_k<<<cdiv(B_ * H_, 256), 256, 0, stream>>>(h, B_ * H_);

  // 2) hoisted xW = x @ Wx  (M=65536, N=128, K=256) — WMMA
  gemm_wmma_bf16<<<gemm_blocks(B_ * T_, A_), 128, 0, stream>>>(
      x, D_, WxT, D_, xW, A_, nullptr, B_ * T_, A_, D_);

  // 3) 22 sequential decode steps
  for (int step = 0; step < STEPS_; ++step) {
    // hWh = h @ Wh  (M=32, N=128, K=256)
    gemm_wmma_bf16<<<gemm_blocks(B_, A_), 128, 0, stream>>>(
        h, H_, WhT, H_, hWh, A_, nullptr, B_, A_, H_);
    // fused scores + softmax + context
    attn_ctx_k<<<B_, 256, 0, stream>>>(xW, hWh, v, x, ctx);
    // GRU input/hidden GEMMs (M=32, N=768, K=256), bias fused
    gemm_wmma_bf16<<<gemm_blocks(B_, 3 * H_), 128, 0, stream>>>(
        ctx, D_, Wihb, D_, gi, 3 * H_, b_ih, B_, 3 * H_, D_);
    gemm_wmma_bf16<<<gemm_blocks(B_, 3 * H_), 128, 0, stream>>>(
        h, H_, Whhb, H_, gh, 3 * H_, b_hh, B_, 3 * H_, H_);
    // gate fusion, h update in place
    gru_gate_k<<<cdiv(B_ * H_, 256), 256, 0, stream>>>(gi, gh, h);
    // logits = h @ W_cls^T + b_cls -> out[:, step, :]  (M=32, N=4367, K=256)
    gemm_wmma_bf16<<<gemm_blocks(B_, C_), 128, 0, stream>>>(
        h, H_, Wclb, H_, out + (size_t)step * C_, STEPS_ * C_, b_cls, B_, C_, H_);
  }
}
